// BilstmCrfStringBatch_24292335026707
// MI455X (gfx1250) — matrix operand order repaired and checked by reading, not compiled, not run on hardware
//
#include <hip/hip_runtime.h>
#include <hip/hip_bf16.h>
#include <math.h>

// ---------------------------------------------------------------------------
// BiLSTM-CRF forward for gfx1250 (MI455X).
//   V=100000 E=128 H=256 (H2=128) T=7 B=64 S=512, START=5 STOP=6
// ---------------------------------------------------------------------------

typedef __bf16 bf16;
typedef bf16  v16bf __attribute__((ext_vector_type(16)));
typedef float v8f   __attribute__((ext_vector_type(8)));

#define S_LEN   512
#define B_SZ    64
#define E_DIM   128
#define H2      128
#define G4      512               // 4*H2 gate width
#define NTOK    (S_LEN * B_SZ)    // 32768 flat tokens (reshape semantics!)
#define T_TAGS  7
#define START_TAG 5
#define STOP_TAG  6

union Frag { uint4 q[2]; v16bf v; };

__device__ __forceinline__ float sigf(float x) { return 1.0f / (1.0f + __expf(-x)); }
__device__ __forceinline__ float tanh_fast(float x) {
  x = fminf(fmaxf(x, -10.0f), 10.0f);
  float e = __expf(2.0f * x);
  return (e - 1.0f) / (e + 1.0f);
}

// ---------------------------------------------------------------------------
// Pack [W_ih | W_hh] per direction into bf16, rows j=0..511, cols k=0..255.
// Also fuse biases: bias[dir][j] = b_ih[j] + b_hh[j].
// ---------------------------------------------------------------------------
__global__ void prep_kernel(const float* __restrict__ wih_f, const float* __restrict__ whh_f,
                            const float* __restrict__ bih_f, const float* __restrict__ bhh_f,
                            const float* __restrict__ wih_b, const float* __restrict__ whh_b,
                            const float* __restrict__ bih_b, const float* __restrict__ bhh_b,
                            bf16* __restrict__ Wcat, float* __restrict__ bias)
{
  int tid = blockIdx.x * blockDim.x + threadIdx.x;       // 262144 threads
  if (tid < 2 * G4 * 256) {
    int dir = tid >> 17;
    int rem = tid & ((1 << 17) - 1);
    int j = rem >> 8;
    int k = rem & 255;
    const float* wih = dir ? wih_b : wih_f;
    const float* whh = dir ? whh_b : whh_f;
    float v = (k < 128) ? wih[j * 128 + k] : whh[j * 128 + (k - 128)];
    Wcat[tid] = (bf16)v;
  }
  if (tid < 2 * G4) {
    int dir = tid >> 9; int j = tid & 511;
    bias[tid] = dir ? (bih_b[j] + bhh_b[j]) : (bih_f[j] + bhh_f[j]);
  }
}

// ---------------------------------------------------------------------------
// Embedding gather -> bf16, flat token order (reshape (B,S,E)->(S,B,E)).
// ---------------------------------------------------------------------------
__global__ void gather_kernel(const int* __restrict__ sent, const float* __restrict__ emb,
                              bf16* __restrict__ X)
{
  int tid = blockIdx.x * blockDim.x + threadIdx.x;       // 131072 threads
  int row = tid >> 2, part = tid & 3;
  if (row >= NTOK) return;
  const float* src = emb + (size_t)sent[row] * E_DIM + part * 32;
  bf16* dst = X + (size_t)row * E_DIM + part * 32;
#pragma unroll
  for (int i = 0; i < 32; ++i) dst[i] = (bf16)src[i];
}

// ---------------------------------------------------------------------------
// Persistent LSTM kernel: one block per direction, 8 waves (wave32).
// Per step: Z(64x512) = [x_t | h](64x256) @ Wcat^T via WMMA bf16 -> f32.
// Wave w owns hidden block u in [16w,16w+16): computes gate tiles i/f/g/o.
// c state lives in registers; h is ping/pong double-buffered bf16 in LDS.
// ---------------------------------------------------------------------------
__global__ __launch_bounds__(256, 1) void lstm_kernel(const bf16* __restrict__ X,
    const bf16* __restrict__ Wcat, const float* __restrict__ bias,
    const float* __restrict__ h0, const float* __restrict__ c0,
    float* __restrict__ hout)
{
  const int dir  = blockIdx.x;
  const int tid  = threadIdx.x;
  const int wav  = tid >> 5;
  const int lane = tid & 31;
  const int n    = lane & 15;
  const int hlf  = lane >> 4;
  const int u    = wav * 16 + n;

  // cols [0,128) = x_t ; [128,256) = h ping0 ; [256,384) = h ping1 ; pad to 392
  __shared__ bf16 A[64][392];

  const bf16*  Wd    = Wcat + (size_t)dir * G4 * 256;
  const float* biasd = bias + dir * G4;
  float*       houtd = hout + (size_t)dir * NTOK * H2;

  // init h (ping=0 region)
  {
    const float* h0d = h0 + dir * B_SZ * H2;
    for (int e = tid; e < B_SZ * H2; e += 256) {
      int b = e >> 7, uu = e & 127;
      A[b][128 + uu] = (bf16)h0d[e];
    }
  }
  // c state in registers: element (b,u) fixed to this lane forever
  float cst[4][8];
  {
    const float* c0d = c0 + dir * B_SZ * H2;
#pragma unroll
    for (int mt = 0; mt < 4; ++mt)
#pragma unroll
      for (int r = 0; r < 8; ++r)
        cst[mt][r] = c0d[(16 * mt + r + 8 * hlf) * H2 + u];
  }
  const float bi  = biasd[0 * H2 + u];
  const float bff = biasd[1 * H2 + u];
  const float bg  = biasd[2 * H2 + u];
  const float bo  = biasd[3 * H2 + u];

  const int xrow = tid >> 2;
  const int xchk = tid & 3;

  int ping = 0;
  for (int it = 0; it < S_LEN; ++it) {
    const int s = dir ? (S_LEN - 1 - it) : it;

    // stage x_t into LDS (64 rows x 128 bf16 = 16KB), prefetch next step
    {
      const uint4* src = (const uint4*)(X + ((size_t)(s * B_SZ + xrow) * E_DIM + xchk * 32));
      uint4* dst = (uint4*)&A[xrow][xchk * 32];
      dst[0] = src[0]; dst[1] = src[1]; dst[2] = src[2]; dst[3] = src[3];
      if (it + 1 < S_LEN) {
        const int s2 = dir ? (s - 1) : (s + 1);
        __builtin_prefetch(X + ((size_t)(s2 * B_SZ + xrow) * E_DIM + xchk * 32), 0, 1);
      }
    }
    __syncthreads();

    const int hwr = 128 + 128 * (ping ^ 1);  // h write region for this step

    for (int mt = 0; mt < 4; ++mt) {
      v8f acc[4] = {};
      const int arow = 16 * mt + n;          // A-matrix M = lane%16
#pragma unroll
      for (int kb = 0; kb < 8; ++kb) {
        // A fragment (16-bit 16x32 layout): two 16B chunks per lane
        Frag fa;
        int col = 32 * kb + 8 * hlf + ((kb >= 4) ? 128 * ping : 0);
        fa.q[0] = *(const uint4*)&A[arow][col];
        fa.q[1] = *(const uint4*)&A[arow][col + 16];
#pragma unroll
        for (int g = 0; g < 4; ++g) {
          // B fragment: B[k][n] = Wcat[j=16*(8g+wav)+n][k], contiguous in k
          Frag fb;
          const bf16* wr = Wd + (size_t)(16 * (g * 8 + wav) + n) * 256 + 32 * kb + 8 * hlf;
          fb.q[0] = *(const uint4*)wr;
          fb.q[1] = *(const uint4*)(wr + 16);
          acc[g] = __builtin_amdgcn_wmma_f32_16x16x32_bf16(
              false, fa.v, false, fb.v, (short)0, acc[g], false, false);
        }
      }
      // gate math + state update for this M-tile (C/D layout: VGPR r -> M=r+8*hlf)
#pragma unroll
      for (int r = 0; r < 8; ++r) {
        const int b = 16 * mt + r + 8 * hlf;
        float zi = acc[0][r] + bi;
        float zf = acc[1][r] + bff;
        float zg = acc[2][r] + bg;
        float zo = acc[3][r] + bo;
        float c  = sigf(zf) * cst[mt][r] + sigf(zi) * tanh_fast(zg);
        cst[mt][r] = c;
        float h  = sigf(zo) * tanh_fast(c);
        A[b][hwr + u] = (bf16)h;                               // next step's input
        houtd[(size_t)(s * B_SZ + b) * H2 + u] = h;            // for feats
      }
    }
    __syncthreads();
    ping ^= 1;
  }
}

// ---------------------------------------------------------------------------
// feats[row][t] = b_out[t] + hf[row]·w_out[t,0:128] + hb[row]·w_out[t,128:256]
// stored with stride 8 (col 7 zero-padded). f32 for CRF accuracy.
// ---------------------------------------------------------------------------
__global__ void feats_kernel(const float* __restrict__ hf, const float* __restrict__ hb,
                             const float* __restrict__ w_out, const float* __restrict__ b_out,
                             float* __restrict__ feats)
{
  int row = blockIdx.x * blockDim.x + threadIdx.x;
  if (row >= NTOK) return;
  const float* xf = hf + (size_t)row * H2;
  const float* xb = hb + (size_t)row * H2;
#pragma unroll 1
  for (int t = 0; t < T_TAGS; ++t) {
    const float* w = w_out + t * 256;
    float acc = b_out[t];
    for (int k = 0; k < 128; ++k) acc += xf[k] * w[k];
    for (int k = 0; k < 128; ++k) acc += xb[k] * w[128 + k];
    feats[(size_t)row * 8 + t] = acc;
  }
  feats[(size_t)row * 8 + 7] = 0.0f;
}

// ---------------------------------------------------------------------------
// CRF forward score + gold score; out[0] = mean(fwd - gold).
// One thread per batch element; trans[i*7+j]: i=prev (scan), gold uses [next][prev].
// ---------------------------------------------------------------------------
__global__ void crf_kernel(const float* __restrict__ feats, const float* __restrict__ trans,
                           const int* __restrict__ tags, float* __restrict__ out)
{
  __shared__ float tr[49];
  __shared__ float red[64];
  const int t = threadIdx.x;
  if (t < 49) tr[t] = trans[t];
  __syncthreads();
  {
    const int b = t;
    float alpha[7];
#pragma unroll
    for (int j = 0; j < 7; ++j) alpha[j] = -10000.0f;
    alpha[START_TAG] = 0.0f;
    for (int s = 0; s < S_LEN; ++s) {
      const float* f = feats + (size_t)(s * B_SZ + b) * 8;
      float na[7];
#pragma unroll
      for (int j = 0; j < 7; ++j) {
        float m = -3.4e38f;
#pragma unroll
        for (int i = 0; i < 7; ++i) m = fmaxf(m, alpha[i] + tr[i * 7 + j]);
        float sum = 0.0f;
#pragma unroll
        for (int i = 0; i < 7; ++i) sum += __expf(alpha[i] + tr[i * 7 + j] - m);
        na[j] = m + __logf(sum) + f[j];
      }
#pragma unroll
      for (int j = 0; j < 7; ++j) alpha[j] = na[j];
    }
    float m = -3.4e38f;
#pragma unroll
    for (int j = 0; j < 7; ++j) m = fmaxf(m, alpha[j] + tr[STOP_TAG * 7 + j]);
    float sum = 0.0f;
#pragma unroll
    for (int j = 0; j < 7; ++j) sum += __expf(alpha[j] + tr[STOP_TAG * 7 + j] - m);
    float fwd = m + __logf(sum);

    int prev = START_TAG;
    float gold = 0.0f;
    for (int s = 0; s < S_LEN; ++s) {
      int nt = tags[b * S_LEN + s];
      gold += tr[nt * 7 + prev] + feats[(size_t)(s * B_SZ + b) * 8 + nt];
      prev = nt;
    }
    gold += tr[STOP_TAG * 7 + prev];
    red[b] = fwd - gold;
  }
  __syncthreads();
  if (t == 0) {
    float s = 0.0f;
    for (int i = 0; i < 64; ++i) s += red[i];
    out[0] = s / (float)B_SZ;
  }
}

// ---------------------------------------------------------------------------
extern "C" void kernel_launch(void* const* d_in, const int* in_sizes, int n_in,
                              void* d_out, int out_size, void* d_ws, size_t ws_size,
                              hipStream_t stream)
{
  const int*   sentence = (const int*)  d_in[0];
  const int*   tags     = (const int*)  d_in[1];
  const float* h0       = (const float*)d_in[2];
  const float* c0       = (const float*)d_in[3];
  const float* emb      = (const float*)d_in[4];
  const float* wih_f    = (const float*)d_in[5];
  const float* whh_f    = (const float*)d_in[6];
  const float* bih_f    = (const float*)d_in[7];
  const float* bhh_f    = (const float*)d_in[8];
  const float* wih_b    = (const float*)d_in[9];
  const float* whh_b    = (const float*)d_in[10];
  const float* bih_b    = (const float*)d_in[11];
  const float* bhh_b    = (const float*)d_in[12];
  const float* w_out    = (const float*)d_in[13];
  const float* b_out    = (const float*)d_in[14];
  const float* trans    = (const float*)d_in[15];

  char* ws = (char*)d_ws;
  bf16*  X     = (bf16*)(ws);                 //  8,388,608 B : 32768 x 128 bf16
  bf16*  Wcat  = (bf16*)(ws + 8388608);       //    524,288 B : 2 x 512 x 256 bf16
  float* bias  = (float*)(ws + 8912896);      //      4,096 B : 2 x 512 f32
  float* hout  = (float*)(ws + 8916992);      // 33,554,432 B : 2 x 32768 x 128 f32
  float* feats = (float*)(ws + 42471424);     //  1,048,576 B : 32768 x 8 f32

  prep_kernel  <<<1024, 256, 0, stream>>>(wih_f, whh_f, bih_f, bhh_f,
                                          wih_b, whh_b, bih_b, bhh_b, Wcat, bias);
  gather_kernel<<< 512, 256, 0, stream>>>(sentence, emb, X);
  lstm_kernel  <<<   2, 256, 0, stream>>>(X, Wcat, bias, h0, c0, hout);
  feats_kernel <<<NTOK / 256, 256, 0, stream>>>(hout, hout + (size_t)NTOK * H2,
                                                w_out, b_out, feats);
  crf_kernel   <<<   1,  64, 0, stream>>>(feats, trans, tags, (float*)d_out);
}
